// Model_46918222742100
// MI455X (gfx1250) — matrix operand order repaired and checked by reading, not compiled, not run on hardware
//
#include <hip/hip_runtime.h>

// ---------------------------------------------------------------------------
// Types / helpers
// ---------------------------------------------------------------------------
typedef unsigned short u16t;
typedef __attribute__((ext_vector_type(16))) __bf16 v16bf;
typedef __attribute__((ext_vector_type(8)))  float  v8f;

union Frag { uint4 u[2]; v16bf v; };

__device__ __forceinline__ float bf2f(u16t v) {
  return __uint_as_float(((unsigned int)v) << 16);
}
__device__ __forceinline__ u16t f2bf(float f) {
  unsigned int u = __float_as_uint(f);
  u += 0x7FFFu + ((u >> 16) & 1u);          // round-to-nearest-even
  return (u16t)(u >> 16);
}
__device__ __forceinline__ float gelu_exact(float x) {
  return 0.5f * x * (1.0f + erff(x * 0.70710678118f));
}

// LDS byte address = low 32 bits of the flat shared-aperture address (ISA 10.2)
__device__ __forceinline__ unsigned lds_addr32(const void* p) {
  return (unsigned)(unsigned long long)p;
}
// CDNA5 async global->LDS copy, 16B per lane, tracked by ASYNCcnt
__device__ __forceinline__ void async_copy_b128(unsigned lds, const void* gaddr) {
  asm volatile("global_load_async_to_lds_b128 %0, %1, off"
               :: "v"(lds), "v"((unsigned long long)gaddr) : "memory");
}
__device__ __forceinline__ void wait_async0() {
  asm volatile("s_wait_asynccnt 0x0" ::: "memory");
}

#define FB_BIAS   1
#define FB_RELU   2
#define FB_GELU   4
#define FB_RES    8
#define FB_ACC    16
#define FB_F32OUT 32
#define FB_TRANSB 64

// ---------------------------------------------------------------------------
// Generic batched bf16 WMMA GEMM:  C = act(alpha*A@B + bias + res (+ Cprev))
// A: [M,K] bf16 row-major (lda).  B: [N,K] (transB=1, preferred: async path)
// or [K,N] (transB=0, LDS-transpose fallback).
// Block tile 128x128x32, 8 waves (wave32), wave tile 32x64 (2x4 fragments).
// Double-buffered LDS; async global->LDS copies overlap with WMMA.
// `full` flags are BLOCK-UNIFORM -> scalar branch, no exec-mask divergence.
// ---------------------------------------------------------------------------
struct GemmP {
  const u16t* A; const u16t* B; void* C;
  const float* bias; const u16t* res;
  int M, N, K, lda, ldb, ldc, ldres, flags;
  float alpha;
  int batInner;
  long long sAo, sAi, sBo, sBi, sCo, sCi, sRo, sRi;
};

// rows x 32 bf16 tile from row-major src -> LDS (80B pitch rows).
// full (uniform): pure async path.  Edge: branchless clamped scalar loads.
__device__ __forceinline__ void load_rm_tile(const u16t* S, int ld, int rows, int K,
                                             int r0, int kt, unsigned int* buf, int t,
                                             bool full) {
  const int r = t >> 1, hc = t & 1;
  const int gr = r0 + r, gk = kt + hc * 16;
  unsigned int* drow = &buf[r * 20 + hc * 8];
  if (full) {
    const u16t* g = S + (long long)gr * ld + gk;
    async_copy_b128(lds_addr32(drow), g);
    async_copy_b128(lds_addr32(drow + 4), g + 8);
  } else {
    const int grc = gr < rows ? gr : rows - 1;
    u16t* d = (u16t*)drow;
#pragma unroll
    for (int e = 0; e < 16; ++e) {
      const int kk = gk + e;
      const int kkc = kk < K ? kk : K - 1;
      const u16t v = S[(long long)grc * ld + kkc];
      d[e] = (gr < rows && kk < K) ? v : (u16t)0;       // v_cndmask, no branches
    }
  }
}

// [K][N] tile -> Bs[n][k] via VGPR + scalar LDS transpose (activation B only)
__device__ __forceinline__ void load_kn_tile(const u16t* B, int ldb, int K, int N,
                                             int n0, int kt, unsigned int* buf, int t,
                                             bool full) {
  const int kw = t >> 3, nb = (t & 7) * 16;
  const int gk = kt + kw;
  const int gkc = gk < K ? gk : K - 1;
  alignas(16) u16t tmp[16];
  if (full) {
    const uint4* sv = (const uint4*)(B + (long long)gk * ldb + n0 + nb);
    ((uint4*)tmp)[0] = sv[0]; ((uint4*)tmp)[1] = sv[1];
  } else {
#pragma unroll
    for (int e = 0; e < 16; ++e) {
      const int gn = n0 + nb + e;
      const int gnc = gn < N ? gn : N - 1;
      const u16t v = B[(long long)gkc * ldb + gnc];
      tmp[e] = (gk < K && gn < N) ? v : (u16t)0;
    }
  }
  u16t* bsh = (u16t*)buf;
#pragma unroll
  for (int e = 0; e < 16; ++e) bsh[(nb + e) * 40 + kw] = tmp[e];
}

__global__ __launch_bounds__(256) void k_gemm(GemmP p) {
  __shared__ unsigned int As[2][128 * 20];   // 128 rows x 32 bf16 (+pad) = 80B/row
  __shared__ unsigned int Bs[2][128 * 20];   // stored as [n][k]

  const int t = threadIdx.x;
  const int bz = blockIdx.z;
  const int bo = bz / p.batInner, bi = bz % p.batInner;
  const u16t* A = p.A + bo * p.sAo + bi * p.sAi;
  const u16t* B = p.B + bo * p.sBo + bi * p.sBi;
  const u16t* R = p.res ? (p.res + bo * p.sRo + bi * p.sRi) : (const u16t*)0;
  const long long cOff = bo * p.sCo + bi * p.sCi;

  const int m0 = blockIdx.y * 128, n0 = blockIdx.x * 128;
  const int lane = t & 31, wv = t >> 5;
  const int wm = wv >> 1, wn = wv & 1;       // 4x2 wave grid
  const bool bT = (p.flags & FB_TRANSB) != 0;

  // block-uniform fullness (rows direction); K direction checked per tile
  const bool aRowsFull = (m0 + 128 <= p.M);
  const bool bRowsFull = bT ? (n0 + 128 <= p.N) : (n0 + 128 <= p.N);

  v8f acc[2][4];
#pragma unroll
  for (int i = 0; i < 2; ++i)
#pragma unroll
    for (int j = 0; j < 4; ++j)
#pragma unroll
      for (int e = 0; e < 8; ++e) acc[i][j][e] = 0.0f;

  const int nTiles = (p.K + 31) >> 5;

  // prologue: stage tile 0 into buffer 0
  {
    const bool kFull = (32 <= p.K);
    load_rm_tile(A, p.lda, p.M, p.K, m0, 0, &As[0][0], t, aRowsFull && kFull);
    if (bT) load_rm_tile(B, p.ldb, p.N, p.K, n0, 0, &Bs[0][0], t, bRowsFull && kFull);
    else    load_kn_tile(B, p.ldb, p.K, p.N, n0, 0, &Bs[0][0], t, bRowsFull && kFull);
  }
  wait_async0();
  __syncthreads();

  int cur = 0;
  for (int ti = 0; ti < nTiles; ++ti) {
    // stage next tile into the other buffer (async, overlapped with WMMA)
    if (ti + 1 < nTiles) {
      const int ktn = (ti + 1) << 5;
      const bool kFull = (ktn + 32 <= p.K);
      load_rm_tile(A, p.lda, p.M, p.K, m0, ktn, &As[cur ^ 1][0], t, aRowsFull && kFull);
      if (bT) load_rm_tile(B, p.ldb, p.N, p.K, n0, ktn, &Bs[cur ^ 1][0], t, bRowsFull && kFull);
      else    load_kn_tile(B, p.ldb, p.K, p.N, n0, ktn, &Bs[cur ^ 1][0], t, bRowsFull && kFull);
    }

    // compute on current buffer
    {
      const int hA = lane >> 4, l15 = lane & 15;
      Frag a[2], b[4];
#pragma unroll
      for (int i = 0; i < 2; ++i) {
        const int r = wm * 32 + i * 16 + l15;
        a[i].u[0] = *(const uint4*)&As[cur][r * 20 + hA * 4];     // k 0-7 / 8-15
        a[i].u[1] = *(const uint4*)&As[cur][r * 20 + 8 + hA * 4]; // k 16-23 / 24-31
      }
#pragma unroll
      for (int j = 0; j < 4; ++j) {
        const int r = wn * 64 + j * 16 + l15;
        b[j].u[0] = *(const uint4*)&Bs[cur][r * 20 + hA * 4];     // k 0-7 / 8-15
        b[j].u[1] = *(const uint4*)&Bs[cur][r * 20 + 8 + hA * 4]; // k 16-23 / 24-31
      }
#pragma unroll
      for (int i = 0; i < 2; ++i)
#pragma unroll
        for (int j = 0; j < 4; ++j)
          acc[i][j] = __builtin_amdgcn_wmma_f32_16x16x32_bf16(
              false, a[i].v, false, b[j].v, (short)0, acc[i][j], false, false);
    }
    wait_async0();      // this wave's async writes to LDS are done
    __syncthreads();    // everyone's writes visible
    cur ^= 1;
  }

  // ---------------- epilogue -------------------------------------------
  const int l15 = lane & 15, hA = lane >> 4;
#pragma unroll
  for (int i = 0; i < 2; ++i)
#pragma unroll
    for (int j = 0; j < 4; ++j)
#pragma unroll
      for (int v = 0; v < 8; ++v) {
        const int row = m0 + wm * 32 + i * 16 + v + hA * 8;
        const int col = n0 + wn * 64 + j * 16 + l15;
        if (row < p.M && col < p.N) {
          float f = acc[i][j][v] * p.alpha;
          if (p.flags & FB_BIAS) f += p.bias[col];
          if (p.flags & FB_RES)  f += bf2f(R[(long long)row * p.ldres + col]);
          const long long ci = cOff + (long long)row * p.ldc + col;
          if (p.flags & FB_ACC) {
            if (p.flags & FB_F32OUT) f += ((const float*)p.C)[ci];
            else                     f += bf2f(((const u16t*)p.C)[ci]);
          }
          if (p.flags & FB_RELU) f = fmaxf(f, 0.0f);
          if (p.flags & FB_GELU) f = gelu_exact(f);
          if (p.flags & FB_F32OUT) ((float*)p.C)[ci] = f;
          else                     ((u16t*)p.C)[ci] = f2bf(f);
        }
      }
}

// ---------------------------------------------------------------------------
// Small kernels
// ---------------------------------------------------------------------------
__global__ void k_cvt(const float* a, u16t* o, long long n) {
  long long i = (long long)blockIdx.x * 256 + threadIdx.x;
  if (i < n) o[i] = f2bf(a[i]);
}

// fp32 [K][N] -> bf16 [N][K]  (weight pre-transpose so GEMM B is always [N][K])
__global__ void k_cvtT(const float* w, u16t* o, int K, int N) {
  long long i = (long long)blockIdx.x * 256 + threadIdx.x;
  if (i < (long long)K * N) {
    int k = (int)(i % K);
    int n = (int)(i / K);
    o[i] = f2bf(w[(long long)k * N + n]);
  }
}

// conv1_w [1024][512][3] -> out[k][co][ci]   ([N][K] per k-slice)
__global__ void k_cvtconv(const float* w, u16t* o) {
  long long i = (long long)blockIdx.x * 256 + threadIdx.x;
  if (i < 3LL * 512 * 1024) {
    int ci = (int)(i % 512);
    int co = (int)((i / 512) % 1024);
    int k  = (int)(i / (512 * 1024));
    o[i] = f2bf(w[(long long)co * 1536 + ci * 3 + k]);
  }
}

// x [B,L,C] f32 -> xc [B*C, L] bf16
__global__ void k_xc(const float* x, u16t* xc) {
  long long i = (long long)blockIdx.x * 256 + threadIdx.x;
  if (i < 2568LL * 4096) {
    int l = (int)(i & 4095);
    int bc = (int)(i >> 12);
    int b = bc / 321, c = bc - b * 321;
    xc[i] = f2bf(x[((long long)b * 4096 + l) * 321 + c]);
  }
}

__global__ __launch_bounds__(256) void k_ln(const u16t* x, u16t* y,
                                            const float* g, const float* bta, int D) {
  __shared__ float sh[256];
  const long long row = blockIdx.x;
  const u16t* xr = x + row * D;
  u16t* yr = y + row * D;
  float s = 0.f, s2 = 0.f;
  for (int d = threadIdx.x; d < D; d += 256) { float v = bf2f(xr[d]); s += v; s2 += v * v; }
  sh[threadIdx.x] = s; __syncthreads();
  for (int o = 128; o > 0; o >>= 1) { if (threadIdx.x < o) sh[threadIdx.x] += sh[threadIdx.x + o]; __syncthreads(); }
  const float mean = sh[0] / D; __syncthreads();
  sh[threadIdx.x] = s2; __syncthreads();
  for (int o = 128; o > 0; o >>= 1) { if (threadIdx.x < o) sh[threadIdx.x] += sh[threadIdx.x + o]; __syncthreads(); }
  const float var = sh[0] / D - mean * mean;
  const float rs = rsqrtf(var + 1e-5f);
  for (int d = threadIdx.x; d < D; d += 256) {
    float v = bf2f(xr[d]);
    yr[d] = f2bf((v - mean) * rs * g[d] + bta[d]);
  }
}

__global__ __launch_bounds__(256) void k_ln2d(u16t* x, const float* g, const float* bta, int n) {
  __shared__ float sh[256];
  u16t* xb = x + (long long)blockIdx.x * n;
  float s = 0.f, s2 = 0.f;
  for (int i = threadIdx.x; i < n; i += 256) { float v = bf2f(xb[i]); s += v; s2 += v * v; }
  sh[threadIdx.x] = s; __syncthreads();
  for (int o = 128; o > 0; o >>= 1) { if (threadIdx.x < o) sh[threadIdx.x] += sh[threadIdx.x + o]; __syncthreads(); }
  const float mean = sh[0] / n; __syncthreads();
  sh[threadIdx.x] = s2; __syncthreads();
  for (int o = 128; o > 0; o >>= 1) { if (threadIdx.x < o) sh[threadIdx.x] += sh[threadIdx.x + o]; __syncthreads(); }
  const float var = sh[0] / n - mean * mean;
  const float rs = rsqrtf(var + 1e-5f);
  __syncthreads();
  for (int i = threadIdx.x; i < n; i += 256) {
    float v = bf2f(xb[i]);
    xb[i] = f2bf((v - mean) * rs * g[i] + bta[i]);
  }
}

__global__ __launch_bounds__(128) void k_softmax(const float* in, u16t* out,
                                                 int W, int ldin, int ldout) {
  __shared__ float sh[128];
  const long long row = blockIdx.x;
  const float* xr = in + row * ldin;
  u16t* yr = out + row * ldout;
  float mx = -1e30f;
  for (int i = threadIdx.x; i < W; i += 128) mx = fmaxf(mx, xr[i]);
  sh[threadIdx.x] = mx; __syncthreads();
  for (int o = 64; o > 0; o >>= 1) { if (threadIdx.x < o) sh[threadIdx.x] = fmaxf(sh[threadIdx.x], sh[threadIdx.x + o]); __syncthreads(); }
  mx = sh[0]; __syncthreads();
  float s = 0.f;
  for (int i = threadIdx.x; i < W; i += 128) s += __expf(xr[i] - mx);
  sh[threadIdx.x] = s; __syncthreads();
  for (int o = 64; o > 0; o >>= 1) { if (threadIdx.x < o) sh[threadIdx.x] += sh[threadIdx.x + o]; __syncthreads(); }
  const float inv = 1.0f / sh[0];
  for (int i = threadIdx.x; i < W; i += 128) yr[i] = f2bf(__expf(xr[i] - mx) * inv);
}

// 14-token attention, one block per sequence, one wave per head (8 heads)
__global__ __launch_bounds__(256) void k_attn14(const u16t* Q, const u16t* K,
                                                const u16t* V, u16t* O) {
  const int bc = blockIdx.x;
  const int h = threadIdx.x >> 5, lane = threadIdx.x & 31;
  const long long base = (long long)bc * 16 * 512 + h * 64;
  for (int i = 0; i < 14; ++i) {
    float s = -1e30f;
    if (lane < 14) {
      float acc = 0.f;
      for (int d = 0; d < 64; ++d)
        acc += bf2f(Q[base + (long long)i * 512 + d]) * bf2f(K[base + (long long)lane * 512 + d]);
      s = acc * 0.125f;      // 1/sqrt(64)
    }
    float mx = s;
    for (int o = 16; o > 0; o >>= 1) mx = fmaxf(mx, __shfl_xor(mx, o, 32));
    const float e = (lane < 14) ? __expf(s - mx) : 0.f;
    float sum = e;
    for (int o = 16; o > 0; o >>= 1) sum += __shfl_xor(sum, o, 32);
    const float pr = e / sum;
    float o0 = 0.f, o1 = 0.f;
    for (int j = 0; j < 14; ++j) {
      const float pj = __shfl(pr, j, 32);
      o0 += pj * bf2f(V[base + (long long)j * 512 + lane]);
      o1 += pj * bf2f(V[base + (long long)j * 512 + lane + 32]);
    }
    O[base + (long long)i * 512 + lane] = f2bf(o0);
    O[base + (long long)i * 512 + lane + 32] = f2bf(o1);
  }
  O[base + 14LL * 512 + lane] = 0; O[base + 14LL * 512 + lane + 32] = 0;
  O[base + 15LL * 512 + lane] = 0; O[base + 15LL * 512 + lane + 32] = 0;
}

// mean over 14 tokens: [bc*16+t, 512] -> [bc, 512]
__global__ __launch_bounds__(256) void k_pool(const u16t* x, u16t* y) {
  const int bc = blockIdx.x;
  for (int d = threadIdx.x; d < 512; d += 256) {
    float s = 0.f;
    for (int t2 = 0; t2 < 14; ++t2) s += bf2f(x[((long long)bc * 16 + t2) * 512 + d]);
    y[(long long)bc * 512 + d] = f2bf(s * (1.0f / 14.0f));
  }
}

__global__ void k_wsum(const u16t* inn, const u16t* whl, const float* wi,
                       const float* ww, u16t* s, long long total) {
  long long i = (long long)blockIdx.x * 256 + threadIdx.x;
  if (i < total) {
    int c = (int)((i >> 9) % 321);
    s[i] = f2bf(wi[c] * bf2f(inn[i]) + ww[c] * bf2f(whl[i]));
  }
}

// [b*321+c, 96] f32 -> out [b, 96, 321]
__global__ void k_outT(const float* m, float* out, long long total) {
  long long i = (long long)blockIdx.x * 256 + threadIdx.x;
  if (i < total) {
    int c = (int)(i % 321);
    long long r = i / 321;
    int pr = (int)(r % 96);
    int b = (int)(r / 96);
    out[i] = m[((long long)b * 321 + c) * 96 + pr];
  }
}

// STFT period branch (16-pt DFT, hann periodic, reflect pad, top-3 logic)
__global__ __launch_bounds__(256) void k_stft(const float* x, int* period) {
  __shared__ int smax[256];
  const int c = blockIdx.x, tid = threadIdx.x;
  const float PI2 = 6.28318530718f;
  int localMax = 0;
  for (int f = tid; f < 513; f += 256) {
    float mag[9];
#pragma unroll
    for (int k = 0; k < 9; ++k) mag[k] = 0.f;
    for (int b = 0; b < 8; ++b) {
      float fr[16];
#pragma unroll
      for (int n = 0; n < 16; ++n) {
        int t = f * 8 - 8 + n;
        if (t < 0) t = -t;
        if (t >= 4096) t = 2 * 4096 - 2 - t;
        const float w = 0.5f - 0.5f * __cosf(PI2 * n * (1.0f / 16.0f));
        fr[n] = x[((long long)b * 4096 + t) * 321 + c] * w;
      }
      for (int k = 1; k < 9; ++k) {
        float re = 0.f, im = 0.f;
#pragma unroll
        for (int n = 0; n < 16; ++n) {
          const float ang = PI2 * k * n * (1.0f / 16.0f);
          re += fr[n] * __cosf(ang);
          im -= fr[n] * __sinf(ang);
        }
        mag[k] += sqrtf(re * re + im * im);
      }
    }
    mag[0] = 0.f;
    int t0 = 0; float v0 = -1.f;
    for (int k = 0; k < 9; ++k) if (mag[k] > v0) { v0 = mag[k]; t0 = k; }
    mag[t0] = -2.f;
    int t1 = 0; float v1 = -1.f;
    for (int k = 0; k < 9; ++k) if (mag[k] > v1) { v1 = mag[k]; t1 = k; }
    mag[t1] = -2.f;
    int t2 = 0; float v2 = -1.f;
    for (int k = 0; k < 9; ++k) if (mag[k] > v2) { v2 = mag[k]; t2 = k; }
    const int chosen = (t0 == 1) ? ((t1 < 1) ? t2 : t1) : t0;
    if (chosen > localMax) localMax = chosen;
  }
  smax[tid] = localMax; __syncthreads();
  for (int s = 128; s > 0; s >>= 1) {
    if (tid < s && smax[tid + s] > smax[tid]) smax[tid] = smax[tid + s];
    __syncthreads();
  }
  if (tid == 0) { int mi = smax[0] > 1 ? smax[0] : 1; period[c] = 4096 / mi; }
}

// ---------------------------------------------------------------------------
// Host-side launch helpers
// ---------------------------------------------------------------------------
static void launch_gemm(hipStream_t st, const u16t* A, int lda, const u16t* B, int ldb,
                        void* C, int ldc, int M, int N, int K, int flags,
                        const float* bias, const u16t* res, int ldres, float alpha,
                        int batO = 1, int batI = 1,
                        long long sAo = 0, long long sAi = 0,
                        long long sBo = 0, long long sBi = 0,
                        long long sCo = 0, long long sCi = 0,
                        long long sRo = 0, long long sRi = 0) {
  GemmP p;
  p.A = A; p.B = B; p.C = C; p.bias = bias; p.res = res;
  p.M = M; p.N = N; p.K = K; p.lda = lda; p.ldb = ldb; p.ldc = ldc; p.ldres = ldres;
  p.flags = flags; p.alpha = alpha; p.batInner = batI;
  p.sAo = sAo; p.sAi = sAi; p.sBo = sBo; p.sBi = sBi;
  p.sCo = sCo; p.sCi = sCi; p.sRo = sRo; p.sRi = sRi;
  dim3 g((unsigned)((N + 127) / 128), (unsigned)((M + 127) / 128), (unsigned)(batO * batI));
  k_gemm<<<g, 256, 0, st>>>(p);
}

extern "C" void kernel_launch(void* const* d_in, const int* in_sizes, int n_in,
                              void* d_out, int out_size, void* d_ws, size_t ws_size,
                              hipStream_t stream) {
  (void)in_sizes; (void)n_in; (void)out_size; (void)ws_size;

  const float* X = (const float*)d_in[0];
  auto F = [&](int i) { return (const float*)d_in[i]; };

  // ---- workspace bump allocator -----------------------------------------
  char* base = (char*)d_ws;
  size_t off = 0;
  auto alloc = [&](size_t elems, size_t esz) -> void* {
    void* p = base + off;
    off += (elems * esz + 255) & ~(size_t)255;
    return p;
  };
  auto allocB = [&](size_t e) { return (u16t*)alloc(e, 2); };
  auto allocF = [&](size_t e) { return (float*)alloc(e, 4); };
  auto cvt = [&](const float* s, u16t* d, long long n) {
    k_cvt<<<(unsigned)((n + 255) / 256), 256, 0, stream>>>(s, d, n);
  };
  auto cvtT = [&](const float* s, u16t* d, int K, int N) {
    long long n = (long long)K * N;
    k_cvtT<<<(unsigned)((n + 255) / 256), 256, 0, stream>>>(s, d, K, N);
  };

  // ---- weight conversion + transpose (once per launch) ------------------
  static const int wK[6] = {512, 512, 512, 512, 512, 2048};
  static const int wN[6] = {512, 512, 512, 512, 2048, 512};
  u16t* wEmb = allocB(4096LL * 512);  cvtT(F(1), wEmb, 4096, 512);   // -> [512][4096]
  u16t* encW[2][6]; u16t* cencW[2][6];
  for (int l = 0; l < 2; ++l) {
    int bI = 3 + l * 16;
    for (int w = 0; w < 6; ++w) {
      encW[l][w] = allocB((size_t)wK[w] * wN[w]);
      cvtT(F(bI + w), encW[l][w], wK[w], wN[w]);
    }
  }
  for (int l = 0; l < 2; ++l) {
    int bI = 47 + l * 16;
    for (int w = 0; w < 6; ++w) {
      cencW[l][w] = allocB((size_t)wK[w] * wN[w]);
      cvtT(F(bI + w), cencW[l][w], wK[w], wN[w]);
    }
  }
  u16t* wVe  = allocB(131072);   cvtT(F(37), wVe, 256, 512);         // -> [512][256]
  u16t* wConv = allocB(1572864);                                     // [k][co][ci]
  k_cvtconv<<<(1572864 + 255) / 256, 256, 0, stream>>>(F(39), wConv);
  u16t* wPc  = allocB(1048576);  cvt(F(41), wPc, 1048576);           // already [co][ci]
  u16t* wPt  = allocB(524288);   cvt(F(43), wPt, 524288);            // already [co][ci]
  u16t* wM1  = allocB(262144);   cvtT(F(81), wM1, 512, 512);
  u16t* wM2  = allocB(262144);   cvtT(F(83), wM2, 512, 512);
  u16t* wL1  = allocB(524288);   cvtT(F(91), wL1, 512, 1024);        // -> [1024][512]
  u16t* wL2  = allocB(1048576);  cvtT(F(93), wL2, 1024, 1024);
  u16t* wL3  = allocB(524288);   cvtT(F(95), wL3, 1024, 512);        // -> [512][1024]
  u16t* wPj  = allocB(49152);    cvtT(F(97), wPj, 512, 96);          // -> [96][512]

  // ---- activation buffers ----------------------------------------------
  const long long ROWS = 41104;                  // 41088 + pad for conv row shift
  u16t* XCb = allocB(2568LL * 4096 + 64);
  u16t* XB = allocB(ROWS * 512);
  u16t* QB = allocB(ROWS * 512);
  u16t* KB = allocB(ROWS * 512);
  u16t* VB = allocB(ROWS * 512);
  u16t* TB = allocB(ROWS * 512);
  u16t* HB = allocB(41088LL * 2048);
  u16t* CB1 = allocB(ROWS * 1024);
  float* SCO = allocF(64LL * 321 * 336);
  u16t* PRB = allocB(64LL * 321 * 336);
  u16t* WHL = allocB(2568LL * 512);
  u16t* INN = allocB(2568LL * 512);
  u16t* SB  = allocB(2568LL * 512);
  u16t* F1  = allocB(2568LL * 1024);
  u16t* F2  = allocB(2568LL * 1024);
  u16t* F3  = allocB(2568LL * 512);
  float* OUTM = allocF(2568LL * 96);

  // ---- independent STFT-period branch -----------------------------------
  k_stft<<<321, 256, 0, stream>>>(X, (int*)((float*)d_out + 8LL * 96 * 321));

  // ---- xc = transpose(x) as bf16 ----------------------------------------
  k_xc<<<(unsigned)((2568LL * 4096 + 255) / 256), 256, 0, stream>>>(X, XCb);

  // =======================================================================
  // Whole-series branch: inverted embedding + 2 encoder layers over C=321
  // =======================================================================
  launch_gemm(stream, XCb, 4096, wEmb, 4096, XB, 512, 2568, 512, 4096,
              FB_BIAS | FB_TRANSB, F(2), nullptr, 0, 1.0f);
  for (int l = 0; l < 2; ++l) {
    const int lb = 3 + l * 16;
    launch_gemm(stream, XB, 512, encW[l][0], 512, QB, 512, 2568, 512, 512,
                FB_BIAS | FB_TRANSB, F(lb + 6), nullptr, 0, 1.0f);
    launch_gemm(stream, XB, 512, encW[l][1], 512, KB, 512, 2568, 512, 512,
                FB_BIAS | FB_TRANSB, F(lb + 7), nullptr, 0, 1.0f);
    launch_gemm(stream, XB, 512, encW[l][2], 512, VB, 512, 2568, 512, 512,
                FB_BIAS | FB_TRANSB, F(lb + 8), nullptr, 0, 1.0f);
    launch_gemm(stream, QB, 512, KB, 512, SCO, 336, 321, 321, 64,
                FB_TRANSB | FB_F32OUT, nullptr, nullptr, 0, 0.125f,
                8, 8, 321LL * 512, 64, 321LL * 512, 64,
                8LL * 321 * 336, 321LL * 336, 0, 0);
    k_softmax<<<64 * 321, 128, 0, stream>>>(SCO, PRB, 321, 336, 336);
    launch_gemm(stream, PRB, 336, VB, 512, TB, 512, 321, 64, 321,
                0, nullptr, nullptr, 0, 1.0f,
                8, 8, 8LL * 321 * 336, 321LL * 336, 321LL * 512, 64,
                321LL * 512, 64, 0, 0);
    launch_gemm(stream, TB, 512, encW[l][3], 512, QB, 512, 2568, 512, 512,
                FB_BIAS | FB_RES | FB_TRANSB, F(lb + 9), XB, 512, 1.0f);
    k_ln<<<2568, 256, 0, stream>>>(QB, XB, F(lb + 12), F(lb + 13), 512);
    launch_gemm(stream, XB, 512, encW[l][4], 512, HB, 2048, 2568, 2048, 512,
                FB_BIAS | FB_GELU | FB_TRANSB, F(lb + 10), nullptr, 0, 1.0f);
    launch_gemm(stream, HB, 2048, encW[l][5], 2048, QB, 512, 2568, 512, 2048,
                FB_BIAS | FB_RES | FB_TRANSB, F(lb + 11), XB, 512, 1.0f);
    k_ln<<<2568, 256, 0, stream>>>(QB, XB, F(lb + 14), F(lb + 15), 512);
  }
  k_ln<<<2568, 256, 0, stream>>>(XB, WHL, F(35), F(36), 512);

  // =======================================================================
  // CTP branch: patch tokens -> conv stack -> encoder(seq 14) -> pool -> MLP
  // =======================================================================
  launch_gemm(stream, XCb, 256, wVe, 256, XB, 512, 41088, 512, 256,
              FB_BIAS | FB_TRANSB, F(38), nullptr, 0, 1.0f);
  for (int k3 = 0; k3 < 3; ++k3) {
    const int fl = ((k3 == 0) ? FB_BIAS : FB_ACC) | FB_TRANSB;
    launch_gemm(stream, XB + k3 * 512, 512, wConv + (long long)k3 * 512 * 1024, 512,
                CB1, 1024, 41088, 1024, 512, fl, F(40), nullptr, 0, 1.0f);
  }
  launch_gemm(stream, CB1, 1024, wPc, 1024, HB, 1024, 41088, 1024, 1024,
              FB_BIAS | FB_TRANSB, F(42), nullptr, 0, 1.0f);
  launch_gemm(stream, HB, 1024, wPt, 1024, TB, 512, 41088, 512, 1024,
              FB_BIAS | FB_TRANSB, F(44), nullptr, 0, 1.0f);
  k_ln<<<41088, 256, 0, stream>>>(TB, XB, F(45), F(46), 512);
  for (int l = 0; l < 2; ++l) {
    const int lb = 47 + l * 16;
    launch_gemm(stream, XB, 512, cencW[l][0], 512, QB, 512, 41088, 512, 512,
                FB_BIAS | FB_TRANSB, F(lb + 6), nullptr, 0, 1.0f);
    launch_gemm(stream, XB, 512, cencW[l][1], 512, KB, 512, 41088, 512, 512,
                FB_BIAS | FB_TRANSB, F(lb + 7), nullptr, 0, 1.0f);
    launch_gemm(stream, XB, 512, cencW[l][2], 512, VB, 512, 41088, 512, 512,
                FB_BIAS | FB_TRANSB, F(lb + 8), nullptr, 0, 1.0f);
    k_attn14<<<2568, 256, 0, stream>>>(QB, KB, VB, TB);
    launch_gemm(stream, TB, 512, cencW[l][3], 512, QB, 512, 41088, 512, 512,
                FB_BIAS | FB_RES | FB_TRANSB, F(lb + 9), XB, 512, 1.0f);
    k_ln<<<41088, 256, 0, stream>>>(QB, XB, F(lb + 12), F(lb + 13), 512);
    launch_gemm(stream, XB, 512, cencW[l][4], 512, HB, 2048, 41088, 2048, 512,
                FB_BIAS | FB_GELU | FB_TRANSB, F(lb + 10), nullptr, 0, 1.0f);
    launch_gemm(stream, HB, 2048, cencW[l][5], 2048, QB, 512, 41088, 512, 2048,
                FB_BIAS | FB_RES | FB_TRANSB, F(lb + 11), XB, 512, 1.0f);
    k_ln<<<41088, 256, 0, stream>>>(QB, XB, F(lb + 14), F(lb + 15), 512);
  }
  k_ln<<<41088, 256, 0, stream>>>(XB, TB, F(79), F(80), 512);
  k_pool<<<2568, 256, 0, stream>>>(TB, QB);
  launch_gemm(stream, QB, 512, wM1, 512, KB, 512, 2568, 512, 512,
              FB_BIAS | FB_RELU | FB_TRANSB, F(82), nullptr, 0, 1.0f);
  launch_gemm(stream, KB, 512, wM2, 512, INN, 512, 2568, 512, 512,
              FB_BIAS | FB_TRANSB, F(84), nullptr, 0, 1.0f);

  // =======================================================================
  // Fusion + head
  // =======================================================================
  k_ln2d<<<8, 256, 0, stream>>>(INN, F(87), F(88), 321 * 512);
  k_ln2d<<<8, 256, 0, stream>>>(WHL, F(89), F(90), 321 * 512);
  k_wsum<<<(unsigned)((2568LL * 512 + 255) / 256), 256, 0, stream>>>(
      INN, WHL, F(85), F(86), SB, 2568LL * 512);
  launch_gemm(stream, SB, 512, wL1, 512, F1, 1024, 2568, 1024, 512,
              FB_BIAS | FB_RELU | FB_TRANSB, F(92), nullptr, 0, 1.0f);
  launch_gemm(stream, F1, 1024, wL2, 1024, F2, 1024, 2568, 1024, 1024,
              FB_BIAS | FB_RELU | FB_TRANSB, F(94), nullptr, 0, 1.0f);
  launch_gemm(stream, F2, 1024, wL3, 1024, F3, 512, 2568, 512, 1024,
              FB_BIAS | FB_GELU | FB_TRANSB, F(96), nullptr, 0, 1.0f);
  launch_gemm(stream, F3, 512, wPj, 512, OUTM, 96, 2568, 96, 512,
              FB_BIAS | FB_F32OUT | FB_TRANSB, F(98), nullptr, 0, 1.0f);
  k_outT<<<(unsigned)((8LL * 96 * 321 + 255) / 256), 256, 0, stream>>>(
      OUTM, (float*)d_out, 8LL * 96 * 321);
}
